// CausalAttention_5815385719336
// MI455X (gfx1250) — matrix operand order repaired and checked by reading, not compiled, not run on hardware
//
#include <hip/hip_runtime.h>
#include <hip/hip_bf16.h>

// ---------------------------------------------------------------------------
// CDNA5 / gfx1250 dual-softmax multi-head attention, bf16 WMMA path with
// Tensor-Data-Mover (TDM) double-buffered LDS staging.
// B=2, S=2048, D=1024, H=16, DK=64.  wave32, 16x16x32 bf16 WMMA.
// ---------------------------------------------------------------------------

typedef __attribute__((ext_vector_type(16))) __bf16 v16bf;
typedef __attribute__((ext_vector_type(8)))  __bf16 v8bf;
typedef __attribute__((ext_vector_type(8)))  float  v8f;
typedef __attribute__((ext_vector_type(4)))  unsigned int u32x4;
typedef __attribute__((ext_vector_type(8)))  int          i32x8;
typedef __attribute__((ext_vector_type(4)))  int          i32x4;

static constexpr int Bb    = 2;
static constexpr int Ss    = 2048;
static constexpr int Dd    = 1024;
static constexpr int Hh    = 16;
static constexpr int DKk   = 64;
static constexpr int Mrows = Bb * Ss;   // 4096

__device__ __forceinline__ v16bf make_frag(v8bf lo, v8bf hi) {
  union { v16bf v; v8bf h[2]; } u;
  u.h[0] = lo; u.h[1] = hi;
  return u.v;
}

// A-fragment (16 rows x 32 k): lanes 0-15 -> M=lane, k elems 0..7 & 16..23;
// lanes 16-31 -> M=lane-16, k elems 8..15 & 24..31.   (ISA 7.12.2)
__device__ __forceinline__ v16bf load_fragA(const __bf16* base, int stride, int lane) {
  const int half = (lane >> 4) & 1;
  const __bf16* p = base + (lane & 15) * stride + half * 8;
  return make_frag(*reinterpret_cast<const v8bf*>(p),
                   *reinterpret_cast<const v8bf*>(p + 16));
}

// B-fragment (32 k x 16 n) from a tile stored [n][k] row-major:
// lanes 0-15 -> n=lane, k 0..7 & 16..23; lanes 16-31 -> n=lane-16, k 8..15 & 24..31.
__device__ __forceinline__ v16bf load_fragB(const __bf16* base, int stride, int row, int lane) {
  const __bf16* p = base + row * stride + ((lane >> 4) & 1) * 8;
  return make_frag(*reinterpret_cast<const v8bf*>(p),
                   *reinterpret_cast<const v8bf*>(p + 16));
}

__device__ __forceinline__ v8f wmma_bf16(v16bf a, v16bf b, v8f c) {
  return __builtin_amdgcn_wmma_f32_16x16x32_bf16(false, a, false, b,
                                                 (short)0, c, false, false);
}

__device__ __forceinline__ float fexp(float x) {            // exp(x) via v_exp_f32
  return __builtin_exp2f(x * 1.4426950408889634f);
}

// ---------------------------------------------------------------------------
// TDM: issue a 2D tensor tile load (global -> LDS).  data_size = 8 bytes; all
// dims/strides in 8-byte units.  D# built per CDNA5 ISA
// (group0: count/lds/global/type, group1: data_size + dims + stride, groups
// 2/3 zero for <=2D tiles).  Issue from ONE wave only; completion tracked by
// TENSORcnt (s_wait_tensorcnt), then published with a workgroup barrier.
// ---------------------------------------------------------------------------
__device__ __forceinline__ void tdm_load_2d(unsigned lds_addr, const void* gaddr,
                                            unsigned tensor_d0, unsigned tensor_d1,
                                            unsigned tile_d0, unsigned tile_d1,
                                            unsigned d0_stride) {
  const unsigned long long ga = (unsigned long long)(uintptr_t)gaddr;
  u32x4 g0;
  g0[0] = 1u;                                            // count=1 valid descriptor
  g0[1] = lds_addr;                                      // lds_addr   [63:32]
  g0[2] = (unsigned)(ga & 0xffffffffu);                  // global lo  [95:64]
  g0[3] = (unsigned)((ga >> 32) & 0x01ffffffu) | (2u << 30); // global hi + type=2
  i32x8 g1;
  g1[0] = (int)(3u << 16);                               // data_size=3 (8B), mask=0
  g1[1] = (int)(tensor_d0 << 16);                        // tensor_dim0[15:0]
  g1[2] = (int)((tensor_d0 >> 16) | (tensor_d1 << 16));  // dim0 hi | dim1 lo
  g1[3] = (int)((tensor_d1 >> 16) | (tile_d0 << 16));    // dim1 hi | tile_dim0
  g1[4] = (int)(tile_d1 & 0xffffu);                      // tile_dim1, tile_dim2=0
  g1[5] = (int)d0_stride;                                // tensor_dim0_stride lo
  g1[6] = 0;                                             // stride hi | dim1_stride
  g1[7] = 0;
  i32x4 z4 = {0, 0, 0, 0};
#if __clang_major__ >= 23
  i32x8 z8 = {0, 0, 0, 0, 0, 0, 0, 0};
  __builtin_amdgcn_tensor_load_to_lds(g0, g1, z4, z4, z8, 0);
#else
  __builtin_amdgcn_tensor_load_to_lds(g0, g1, z4, z4, 0);
#endif
}

__device__ __forceinline__ unsigned lds_off(const void* p) {
  return (unsigned)(uintptr_t)p;   // low 32 bits of a shared pointer = LDS addr
}

// ---------------------------------------------------------------------------
// Stage 1: f32 -> bf16 cast
// ---------------------------------------------------------------------------
__global__ void cast_f32_to_bf16(const float* __restrict__ src,
                                 __bf16* __restrict__ dst, int n) {
  int i = blockIdx.x * blockDim.x + threadIdx.x;
  if (i < n) dst[i] = (__bf16)src[i];
}

// ---------------------------------------------------------------------------
// Stage 2/4: Y[m,n] = sum_k X[m,k] * W[n,k] + bias[n]    (torch Linear)
// M=4096, N=1024, K=1024.  256 threads = 8 waves; block tile 128x256; wave
// tile 64x64 (4x4 WMMA tiles -> 16 WMMAs per K-step against 8 fragment
// loads); K-step 32; TDM double-buffered A/W tiles (48 KB LDS).
// MODE 0: f32 out [m][n]      (final projections -> d_out)
// MODE 1: bf16 out [b,h,s,dk] (Q, K)
// MODE 2: bf16 out [b,h,dk,s] (V transposed -> contiguous PV B-fragments)
// ---------------------------------------------------------------------------
template <int MODE>
__global__ __launch_bounds__(256) void linear_bf16_kernel(
    const __bf16* __restrict__ X,    // [4096,1024]
    const __bf16* __restrict__ W,    // [1024,1024] (row n holds W[n,:])
    const float*  __restrict__ bias, // [1024]
    void* __restrict__ out) {
  __shared__ __align__(16) __bf16 As[2][128 * 32];   // 16 KB
  __shared__ __align__(16) __bf16 Ws[2][256 * 32];   // 32 KB

  const int t = threadIdx.x, lane = t & 31, w = t >> 5;
  const int mbase = blockIdx.y * 128, nbase = blockIdx.x * 256;
  const int wm = (w >> 2) * 64, wn = (w & 3) * 64;

  v8f zero = {};
  v8f acc[4][4];
#pragma unroll
  for (int i = 0; i < 4; i++)
#pragma unroll
    for (int j = 0; j < 4; j++) acc[i][j] = zero;

  const __bf16* Xb = X + (size_t)mbase * Dd;   // block's A rows
  const __bf16* Wb = W + (size_t)nbase * Dd;   // block's W rows
  constexpr int NT = Dd / 32;                  // 32 K-tiles
  // 2D tiles: 32 cols * 2B = 64B = 8 units per row; stride 1024*2/8 = 256.
  if (w == 0) {
    tdm_load_2d(lds_off(&As[0][0]), Xb, 8, 4096, 8, 128, 256);
    tdm_load_2d(lds_off(&Ws[0][0]), Wb, 8, 1024, 8, 256, 256);
  }
  for (int tile = 0; tile < NT; ++tile) {
    const int p = tile & 1;
    if (w == 0) {
      if (tile + 1 < NT) {
        tdm_load_2d(lds_off(&As[1 - p][0]), Xb + (tile + 1) * 32, 8, 4096, 8, 128, 256);
        tdm_load_2d(lds_off(&Ws[1 - p][0]), Wb + (tile + 1) * 32, 8, 1024, 8, 256, 256);
        __builtin_amdgcn_s_wait_tensorcnt(2);  // current tile's pair done
      } else {
        __builtin_amdgcn_s_wait_tensorcnt(0);
      }
    }
    __syncthreads();                           // publish buffers[p]

    v16bf bfr[4];
#pragma unroll
    for (int j = 0; j < 4; j++)
      bfr[j] = load_fragB(Ws[p], 32, wn + j * 16 + (lane & 15), lane);
#pragma unroll
    for (int i = 0; i < 4; i++) {
      v16bf afr = load_fragA(&As[p][(wm + i * 16) * 32], 32, lane);
#pragma unroll
      for (int j = 0; j < 4; j++)
        acc[i][j] = wmma_bf16(afr, bfr[j], acc[i][j]);
    }
    __syncthreads();                           // done reading buffers[p]
  }

  const int col = lane & 15, rofs = (lane >> 4) * 8;
#pragma unroll
  for (int i = 0; i < 4; i++) {
#pragma unroll
    for (int j = 0; j < 4; j++) {
      const int n = nbase + wn + j * 16 + col;
      const float bn = bias[n];
#pragma unroll
      for (int r = 0; r < 8; r++) {
        const int m = mbase + wm + i * 16 + rofs + r;
        const float val = acc[i][j][r] + bn;
        if constexpr (MODE == 0) {
          reinterpret_cast<float*>(out)[(size_t)m * Dd + n] = val;
        } else if constexpr (MODE == 1) {
          const int bi = m >> 11, s = m & 2047, hi = n >> 6, dk = n & 63;
          reinterpret_cast<__bf16*>(out)[(((size_t)(bi * Hh + hi) * Ss) + s) * DKk + dk] = (__bf16)val;
        } else {
          const int bi = m >> 11, s = m & 2047, hi = n >> 6, dk = n & 63;
          reinterpret_cast<__bf16*>(out)[(((size_t)(bi * Hh + hi) * DKk) + dk) * Ss + s] = (__bf16)val;
        }
      }
    }
  }
}

// ---------------------------------------------------------------------------
// Stage 3: dual-branch flash attention.
// Grid (S/128, H, B), 256 threads = 8 waves; wave owns 16 q rows.
// TDM double-buffered K/Vt tiles; per 32-col K-step: 4 score WMMAs, online
// softmax of +s and -s (row max/min via shuffles; row SUMS via an extra WMMA
// against an all-ones B matrix -> l lives in a C-fragment, no shuffles),
// P through per-wave LDS (C-layout -> A-layout), 8 PV WMMAs + 2 l-WMMAs.
// ---------------------------------------------------------------------------
__global__ __launch_bounds__(256) void attn_dual_kernel(
    const __bf16* __restrict__ Q,    // [B,H,S,DK]
    const __bf16* __restrict__ Km,   // [B,H,S,DK]
    const __bf16* __restrict__ Vt,   // [B,H,DK,S]
    __bf16* __restrict__ Oneg,       // [B,S,D]  softmax(-s) branch
    __bf16* __restrict__ Opos) {     // [B,S,D]  softmax(+s) branch
  __shared__ __align__(16) __bf16 ldsK[2][32 * 64];     // K tile [s=32][d=64]
  __shared__ __align__(16) __bf16 ldsV[2][64 * 32];     // Vt tile [dv=64][s=32]
  __shared__ __align__(16) __bf16 ldsP[8][2][16 * 32];  // per-wave P tiles

  const int t = threadIdx.x, lane = t & 31, w = t >> 5;
  const int h = blockIdx.y, b = blockIdx.z;
  const int q0 = blockIdx.x * 128 + w * 16;

  const __bf16* Qb = Q  + (size_t)(b * Hh + h) * Ss * DKk;
  const __bf16* Kb = Km + (size_t)(b * Hh + h) * Ss * DKk;
  const __bf16* Vb = Vt + (size_t)(b * Hh + h) * DKk * Ss;

  // Q fragments for the wave's 16 rows (d 0..31 and 32..63), held in VGPRs.
  const v16bf aq0 = load_fragA(Qb + (size_t)q0 * DKk,      DKk, lane);
  const v16bf aq1 = load_fragA(Qb + (size_t)q0 * DKk + 32, DKk, lane);

  v16bf ones;
#pragma unroll
  for (int i = 0; i < 16; i++) ones[i] = (__bf16)1.0f;

  v8f zero = {};
  v8f accp[4], accn[4], laccp = zero, laccn = zero;
  float mpos[8], mneg[8];
#pragma unroll
  for (int d = 0; d < 4; d++) { accp[d] = zero; accn[d] = zero; }
#pragma unroll
  for (int r = 0; r < 8; r++) { mpos[r] = -1e30f; mneg[r] = -1e30f; }

  __bf16* Pp = ldsP[w][0];
  __bf16* Pn = ldsP[w][1];
  const int ccol = lane & 15, rofs = (lane >> 4) * 8;

  constexpr int NT = Ss / 32;   // 64 K-tiles
  // K tile: rows of 64*2B=128B=16 units, 32 rows, stride 16 (contiguous).
  // V tile: rows of 32*2B= 64B= 8 units, 64 rows, stride S*2/8 = 512.
  if (w == 0) {
    tdm_load_2d(lds_off(&ldsK[0][0]), Kb, 16, Ss, 16, 32, 16);
    tdm_load_2d(lds_off(&ldsV[0][0]), Vb,  8, DKk, 8, 64, 512);
  }

  for (int tile = 0; tile < NT; ++tile) {
    const int p = tile & 1;
    if (w == 0) {
      if (tile + 1 < NT) {
        tdm_load_2d(lds_off(&ldsK[1 - p][0]), Kb + (size_t)(tile + 1) * 32 * DKk, 16, Ss, 16, 32, 16);
        tdm_load_2d(lds_off(&ldsV[1 - p][0]), Vb + (size_t)(tile + 1) * 32,        8, DKk, 8, 64, 512);
        __builtin_amdgcn_s_wait_tensorcnt(2);
      } else {
        __builtin_amdgcn_s_wait_tensorcnt(0);
      }
    }
    __syncthreads();                                     // buffers[p] ready

    // ---- scores: s[16q x 32k] = Q Kt / 8 ------------------------------------
    v8f s0 = zero, s1 = zero;
    {
      v16bf bk00 = load_fragB(&ldsK[p][0],  64,  0 + ccol, lane);  // j 0..15, d 0..31
      v16bf bk01 = load_fragB(&ldsK[p][32], 64,  0 + ccol, lane);  // j 0..15, d 32..63
      v16bf bk10 = load_fragB(&ldsK[p][0],  64, 16 + ccol, lane);  // j 16..31
      v16bf bk11 = load_fragB(&ldsK[p][32], 64, 16 + ccol, lane);
      s0 = wmma_bf16(aq0, bk00, s0); s0 = wmma_bf16(aq1, bk01, s0);
      s1 = wmma_bf16(aq0, bk10, s1); s1 = wmma_bf16(aq1, bk11, s1);
    }
#pragma unroll
    for (int r = 0; r < 8; r++) { s0[r] *= 0.125f; s1[r] *= 0.125f; }

    // ---- online max for both branches (row max/min across 32 cols) ----------
    float apos[8], aneg[8];
#pragma unroll
    for (int r = 0; r < 8; r++) {
      float mx = fmaxf(s0[r], s1[r]);
      float mn = fminf(s0[r], s1[r]);
#pragma unroll
      for (int msk = 1; msk < 16; msk <<= 1) {
        mx = fmaxf(mx, __shfl_xor(mx, msk, 32));
        mn = fminf(mn, __shfl_xor(mn, msk, 32));
      }
      const float mp_new = fmaxf(mpos[r], mx);
      const float mn_new = fmaxf(mneg[r], -mn);
      apos[r] = fexp(mpos[r] - mp_new);
      aneg[r] = fexp(mneg[r] - mn_new);
      mpos[r] = mp_new; mneg[r] = mn_new;
    }

    // ---- probabilities; rescale running accumulators ------------------------
    v8f p0p, p1p, p0n, p1n;
#pragma unroll
    for (int r = 0; r < 8; r++) {
      p0p[r] = fexp( s0[r] - mpos[r]);  p1p[r] = fexp( s1[r] - mpos[r]);
      p0n[r] = fexp(-s0[r] - mneg[r]);  p1n[r] = fexp(-s1[r] - mneg[r]);
      laccp[r] *= apos[r];  laccn[r] *= aneg[r];
    }
#pragma unroll
    for (int d = 0; d < 4; d++)
#pragma unroll
      for (int r = 0; r < 8; r++) { accp[d][r] *= apos[r]; accn[d][r] *= aneg[r]; }

    // ---- P (C-layout) -> per-wave LDS -> A-layout ---------------------------
#pragma unroll
    for (int r = 0; r < 8; r++) {
      Pp[(rofs + r) * 32 + ccol]      = (__bf16)p0p[r];
      Pp[(rofs + r) * 32 + ccol + 16] = (__bf16)p1p[r];
      Pn[(rofs + r) * 32 + ccol]      = (__bf16)p0n[r];
      Pn[(rofs + r) * 32 + ccol + 16] = (__bf16)p1n[r];
    }
    v16bf ap = load_fragA(Pp, 32, lane);
    v16bf an = load_fragA(Pn, 32, lane);
    laccp = wmma_bf16(ap, ones, laccp);        // row sums -> l (no shuffles)
    laccn = wmma_bf16(an, ones, laccn);
#pragma unroll
    for (int dv = 0; dv < 4; dv++) {
      v16bf bv = load_fragB(&ldsV[p][0], 32, dv * 16 + ccol, lane);
      accp[dv] = wmma_bf16(ap, bv, accp[dv]);
      accn[dv] = wmma_bf16(an, bv, accn[dv]);
    }
    __syncthreads();                            // done reading buffers[p]
  }

  // ---- normalize (v_rcp_f32) and write O (bf16, [b,s,d]) --------------------
  float rp[8], rn[8];
#pragma unroll
  for (int r = 0; r < 8; r++) {
    rp[r] = __builtin_amdgcn_rcpf(laccp[r]);
    rn[r] = __builtin_amdgcn_rcpf(laccn[r]);
  }
#pragma unroll
  for (int dv = 0; dv < 4; dv++) {
#pragma unroll
    for (int r = 0; r < 8; r++) {
      const int q = q0 + rofs + r;
      const size_t o = ((size_t)b * Ss + q) * Dd + h * DKk + dv * 16 + ccol;
      Oneg[o] = (__bf16)(accn[dv][r] * rn[r]);
      Opos[o] = (__bf16)(accp[dv][r] * rp[r]);
    }
  }
}

// ---------------------------------------------------------------------------
// Host-side orchestration (graph-capture safe: launches only).
// Workspace layout (bytes), needs ~56 MiB.
// ---------------------------------------------------------------------------
extern "C" void kernel_launch(void* const* d_in, const int* in_sizes, int n_in,
                              void* d_out, int out_size, void* d_ws, size_t ws_size,
                              hipStream_t stream) {
  const float* x  = (const float*)d_in[0];
  const float* Wq = (const float*)d_in[1];  const float* bq = (const float*)d_in[2];
  const float* Wk = (const float*)d_in[3];  const float* bk = (const float*)d_in[4];
  const float* Wv = (const float*)d_in[5];  const float* bv = (const float*)d_in[6];
  const float* Wo = (const float*)d_in[7];  const float* bo = (const float*)d_in[8];

  char* ws = (char*)d_ws;
  const size_t MiB = 1024 * 1024;
  __bf16* xb   = (__bf16*)(ws + 0);
  __bf16* wqb  = (__bf16*)(ws +  8 * MiB);
  __bf16* wkb  = (__bf16*)(ws + 10 * MiB);
  __bf16* wvb  = (__bf16*)(ws + 12 * MiB);
  __bf16* wob  = (__bf16*)(ws + 14 * MiB);
  __bf16* Qb   = (__bf16*)(ws + 16 * MiB);
  __bf16* Kb   = (__bf16*)(ws + 24 * MiB);
  __bf16* Vtb  = (__bf16*)(ws + 32 * MiB);
  __bf16* Oneg = (__bf16*)(ws + 40 * MiB);
  __bf16* Opos = (__bf16*)(ws + 48 * MiB);

  float* out_neg = (float*)d_out;                       // reference `out`
  float* out_pos = out_neg + (size_t)Mrows * Dd;        // reference `out_comp`

  // Stage 1: casts
  {
    const int nx = Mrows * Dd, nw = Dd * Dd, thr = 256;
    cast_f32_to_bf16<<<(nx + thr - 1) / thr, thr, 0, stream>>>(x,  xb,  nx);
    cast_f32_to_bf16<<<(nw + thr - 1) / thr, thr, 0, stream>>>(Wq, wqb, nw);
    cast_f32_to_bf16<<<(nw + thr - 1) / thr, thr, 0, stream>>>(Wk, wkb, nw);
    cast_f32_to_bf16<<<(nw + thr - 1) / thr, thr, 0, stream>>>(Wv, wvb, nw);
    cast_f32_to_bf16<<<(nw + thr - 1) / thr, thr, 0, stream>>>(Wo, wob, nw);
  }

  // Stage 2: QKV projections
  dim3 ggrid(Dd / 256, Mrows / 128);   // (4, 32)
  linear_bf16_kernel<1><<<ggrid, 256, 0, stream>>>(xb, wqb, bq, (void*)Qb);
  linear_bf16_kernel<1><<<ggrid, 256, 0, stream>>>(xb, wkb, bk, (void*)Kb);
  linear_bf16_kernel<2><<<ggrid, 256, 0, stream>>>(xb, wvb, bv, (void*)Vtb);

  // Stage 3: dual-softmax attention
  dim3 agrid(Ss / 128, Hh, Bb);        // (16, 16, 2)
  attn_dual_kernel<<<agrid, 256, 0, stream>>>(Qb, Kb, Vtb, Oneg, Opos);

  // Stage 4: output projections
  linear_bf16_kernel<0><<<ggrid, 256, 0, stream>>>(Oneg, wob, bo, (void*)out_neg);
  linear_bf16_kernel<0><<<ggrid, 256, 0, stream>>>(Opos, wob, bo, (void*)out_pos);
}
